// Model_end_to_end_28587302322242
// MI455X (gfx1250) — compile-verified, matrix operand order checked by reading, not run on hardware
//
#include <hip/hip_runtime.h>
#include <hip/hip_bf16.h>
#include <math.h>

// ---------------- types ----------------
typedef _Float16 half_t;
typedef __attribute__((ext_vector_type(16))) _Float16 v16h;
typedef __attribute__((ext_vector_type(8)))  float    v8f;

#define PI_F 3.14159265358979f
#define DX_C 0.015625f
#define DT_C 0.003f

// ============================================================
// Utility kernels
// ============================================================
__global__ void k_zero_f32(float* p, int n) {
    int i = blockIdx.x * blockDim.x + threadIdx.x;
    if (i < n) p[i] = 0.f;
}

// pack conv weight OIHW f32 -> wT [CoutPad][KPad] f16, K=(ky*3+kx)*CinPad+ci
__global__ void k_pack_w(const float* __restrict__ w, half_t* __restrict__ wT,
                         int Cout, int Cin, int CinPad, int CoutPad, int KPad) {
    int i = blockIdx.x * blockDim.x + threadIdx.x;
    if (i >= CoutPad * KPad) return;
    int co = i / KPad;
    int k  = i - co * KPad;
    int kk = k / CinPad;       // ky*3+kx
    int ci = k - kk * CinPad;
    float v = 0.f;
    if (co < Cout && ci < Cin && kk < 9) {
        int ky = kk / 3, kx = kk - ky * 3;
        v = w[(((size_t)co * Cin + ci) * 3 + ky) * 3 + kx];
    }
    wT[i] = (half_t)v;
}

__global__ void k_pack_vec(const float* __restrict__ src, float* __restrict__ dst,
                           int Creal, int Cpad) {
    int i = threadIdx.x;
    if (i < Cpad) dst[i] = (i < Creal) ? src[i] : 0.f;
}

// ============================================================
// Physics front-end
// ============================================================
// x [32][4][256][256] f32 -> xr [(b*4+ch)][128][128] f32 ; dsum += sum(wx0)
__global__ void k_down(const float* __restrict__ x, float* __restrict__ xr,
                       float* __restrict__ dsum) {
    __shared__ float red[256];
    size_t i = (size_t)blockIdx.x * 256 + threadIdx.x;  // over 32*4*128*128
    int xo = i & 127; size_t t = i >> 7;
    int yo = t & 127; t >>= 7;
    int ch = t & 3;   int b = (int)(t >> 2);
    const float* src = x + ((((size_t)b * 4 + ch) * 256 + yo * 2) * 256 + xo * 2);
    float v = 0.25f * (src[0] + src[1] + src[256] + src[257]);
    xr[i] = v;
    red[threadIdx.x] = (ch == 0) ? v : 0.f;
    __syncthreads();
    for (int s = 128; s > 0; s >>= 1) {
        if ((int)threadIdx.x < s) red[threadIdx.x] += red[threadIdx.x + s];
        __syncthreads();
    }
    if (threadIdx.x == 0) atomicAdd(dsum, red[0]);
}

// forward row DFT of wx0/wy0 -> Fr[sel][b][y][k] (complex)
__global__ void k_rowdft(const float* __restrict__ xr, float2* __restrict__ Fr) {
    __shared__ float row[128];
    int by = blockIdx.x;            // b*128+y
    int sel = blockIdx.y;           // 0: wx, 1: wy (channels 0/1)
    int k = threadIdx.x;
    const float* src = xr + ((((size_t)(by >> 7)) * 4 + sel) * 128 + (by & 127)) * 128;
    row[k] = src[k];
    __syncthreads();
    float re = 0.f, im = 0.f;
    float w = -2.f * PI_F * (float)k / 128.f;
    for (int xx = 0; xx < 128; ++xx) {
        float s, c; __sincosf(w * (float)xx, &s, &c);
        re += row[xx] * c; im += row[xx] * s;
    }
    Fr[((size_t)sel * 32 * 128 + by) * 128 + k] = make_float2(re, im);
}

// column DFT + spectral operator -> hat[b][ky][kx]
__global__ void k_colop(const float2* __restrict__ Fr, float2* __restrict__ hat,
                        const float* __restrict__ dsum) {
    __shared__ float2 cx[128];
    __shared__ float2 cy[128];
    int b = blockIdx.x, kx = blockIdx.y, ky = threadIdx.x;
    cx[ky] = Fr[((size_t)b * 128 + ky) * 128 + kx];
    cy[ky] = Fr[((size_t)32 * 128 * 128) + ((size_t)b * 128 + ky) * 128 + kx];
    __syncthreads();
    float fxr = 0.f, fxi = 0.f, fyr = 0.f, fyi = 0.f;
    float w = -2.f * PI_F * (float)ky / 128.f;
    for (int y = 0; y < 128; ++y) {
        float s, c; __sincosf(w * (float)y, &s, &c);
        fxr += cx[y].x * c - cx[y].y * s;  fxi += cx[y].x * s + cx[y].y * c;
        fyr += cy[y].x * c - cy[y].y * s;  fyi += cy[y].x * s + cy[y].y * c;
    }
    float fqx = (float)((kx < 64) ? kx : kx - 128) / (128.f * DX_C);
    float fqy = (float)((ky < 64) ? ky : ky - 128) / (128.f * DX_C);
    float kxv = 2.f * PI_F * fqx, kyv = 2.f * PI_F * fqy;
    float rad = kxv * kxv + kyv * kyv;
    if (rad == 0.f) rad = 1.f;
    float sre = kxv * fxr + kyv * fyr;
    float sim = kxv * fxi + kyv * fyi;
    float2 hv = make_float2(sim / rad, -sre / rad);   // -i*s / rad
    if (kx == 0 && ky == 0) { hv.x = *dsum; hv.y = 0.f; }
    hat[((size_t)b * 128 + ky) * 128 + kx] = hv;
}

// inverse column DFT -> G[b][y][kx]
__global__ void k_invcol(const float2* __restrict__ hat, float2* __restrict__ G) {
    __shared__ float2 col[128];
    int b = blockIdx.x, kx = blockIdx.y, y = threadIdx.x;
    col[y] = hat[((size_t)b * 128 + y) * 128 + kx];
    __syncthreads();
    float gr = 0.f, gi = 0.f;
    float w = 2.f * PI_F * (float)y / 128.f;
    for (int ky = 0; ky < 128; ++ky) {
        float s, c; __sincosf(w * (float)ky, &s, &c);
        gr += col[ky].x * c - col[ky].y * s;
        gi += col[ky].x * s + col[ky].y * c;
    }
    G[((size_t)b * 128 + y) * 128 + kx] = make_float2(gr * (1.f / 128.f), gi * (1.f / 128.f));
}

// inverse row DFT (real part) -> u[b][y][x]
__global__ void k_invrow(const float2* __restrict__ G, float* __restrict__ u) {
    __shared__ float2 row[128];
    int by = blockIdx.x;   // b*128+y
    int xx = threadIdx.x;
    row[xx] = G[(size_t)by * 128 + xx];
    __syncthreads();
    float acc = 0.f;
    float w = 2.f * PI_F * (float)xx / 128.f;
    for (int kx = 0; kx < 128; ++kx) {
        float s, c; __sincosf(w * (float)kx, &s, &c);
        acc += row[kx].x * c - row[kx].y * s;
    }
    u[(size_t)by * 128 + xx] = acc * (1.f / 128.f);
}

// 20 Verlet steps fully in LDS (256KB per WGP; CDNA5 has 320KB/WGP).
// u plane is pulled HBM->LDS with ASYNCcnt-tracked async loads.
// Writes prop f16 [b][y][x][16] = (wx,wy,wtc,vel,0..)
__global__ void __launch_bounds__(1024)
k_verlet(const float* __restrict__ u_in, const float* __restrict__ xr,
         half_t* __restrict__ prop) {
    extern __shared__ float sm[];
    float* u  = sm;
    float* u1 = sm + 16384;
    float* ut = sm + 2 * 16384;
    float* c2 = sm + 3 * 16384;
    int b = blockIdx.x, tid = threadIdx.x;
    // ---- async HBM -> LDS fill of u (64KB, 4 rounds of b128/lane) ----
    {
        unsigned uLds = (unsigned)(size_t)(void*)u;
        int gbase = b * 65536;                 // byte offset of image b in u_in
        for (int r = 0; r < 4; ++r) {
            int off = (tid + r * 1024) * 16;   // bytes
            asm volatile("global_load_async_to_lds_b128 %0, %1, %2 offset:0"
                         :: "v"(uLds + (unsigned)off), "v"(gbase + off), "s"(u_in)
                         : "memory");
        }
    }
    const float* wtc0 = xr + ((size_t)b * 4 + 2) * 16384;
    const float* vel  = xr + ((size_t)b * 4 + 3) * 16384;
    for (int i = tid; i < 16384; i += 1024) {
        float v = vel[i];
        ut[i] = wtc0[i] * v;
        c2[i] = v * v;
    }
    asm volatile("s_wait_asynccnt 0" ::: "memory");
    __syncthreads();
    const float idx2 = 1.f / (DX_C * DX_C);
    for (int stp = 0; stp < 20; ++stp) {
        for (int i = tid; i < 16384; i += 1024) {
            int y = i >> 7, x = i & 127;
            int xp = (x + 1) & 127, xm = (x + 127) & 127;
            int yp = ((y + 1) & 127) << 7, ym = ((y + 127) & 127) << 7;
            float lap0 = (u[(y << 7) | xp] + u[(y << 7) | xm] + u[yp | x] + u[ym | x]
                          - 4.f * u[i]) * idx2;
            u1[i] = u[i] + DT_C * ut[i] + 0.5f * DT_C * DT_C * c2[i] * lap0;
        }
        __syncthreads();
        for (int i = tid; i < 16384; i += 1024) {
            int y = i >> 7, x = i & 127;
            int xp = (x + 1) & 127, xm = (x + 127) & 127;
            int yp = ((y + 1) & 127) << 7, ym = ((y + 127) & 127) << 7;
            float lap0 = (u[(y << 7) | xp] + u[(y << 7) | xm] + u[yp | x] + u[ym | x]
                          - 4.f * u[i]) * idx2;
            float lap1 = (u1[(y << 7) | xp] + u1[(y << 7) | xm] + u1[yp | x] + u1[ym | x]
                          - 4.f * u1[i]) * idx2;
            ut[i] += 0.5f * DT_C * c2[i] * (lap0 + lap1);
        }
        __syncthreads();
        for (int i = tid; i < 16384; i += 1024) u[i] = u1[i];
        __syncthreads();
    }
    for (int i = tid; i < 16384; i += 1024) {
        int y = i >> 7, x = i & 127;
        float wx = (x < 127) ? (u[i + 1]   - u[i]) * (1.f / DX_C) : 0.f;
        float wy = (y < 127) ? (u[i + 128] - u[i]) * (1.f / DX_C) : 0.f;
        float v = sqrtf(c2[i]);
        size_t base = ((size_t)b * 16384 + i) * 16;
        prop[base + 0] = (half_t)wx;
        prop[base + 1] = (half_t)wy;
        prop[base + 2] = (half_t)(ut[i] / v);
        prop[base + 3] = (half_t)v;
        for (int c = 4; c < 16; ++c) prop[base + c] = (half_t)0.f;
    }
}

// ============================================================
// Implicit-GEMM 3x3 conv via WMMA f16 (acc f32).
// - K extent compile-time (CINPAD in {16,32}), chunk loop fully unrolled.
// - Per wave: 32-pixel x 16-cout tile -> 2 WMMAs share one B fragment.
// - Layer weights staged once per block into LDS via async b128 loads.
// act: f16 [B][H][W][CINPAD]; wT: f16 [CoutPad][KPAD]; out: f32 [pix][CoutPad]
// ============================================================
template <int CINPAD>
__global__ void __launch_bounds__(256)
k_conv_wmma(const half_t* __restrict__ act, const half_t* __restrict__ wT,
            const float* __restrict__ bias, float* __restrict__ out,
            float* __restrict__ stats,
            int H, int W, int Ho, int Wo, int CoutPad, int stride) {
    constexpr int KPAD = ((9 * CINPAD) + 31) & ~31;
    constexpr int NCH  = KPAD / 32;
    __shared__ __align__(32) half_t ldsW[32 * KPAD];      // <=18KB
    __shared__ __align__(32) half_t ldsA[8][32 * 32];     // 16KB

    // ---- async stage of the whole layer weight matrix into LDS ----
    {
        int nbytes = CoutPad * KPAD * 2;
        unsigned wLds = (unsigned)(size_t)(void*)&ldsW[0];
        for (int off = (int)threadIdx.x * 16; off < nbytes; off += 256 * 16) {
            asm volatile("global_load_async_to_lds_b128 %0, %1, %2 offset:0"
                         :: "v"(wLds + (unsigned)off), "v"(off), "s"(wT)
                         : "memory");
        }
        asm volatile("s_wait_asynccnt 0" ::: "memory");
    }
    __syncthreads();

    int wave = threadIdx.x >> 5;
    int lane = threadIdx.x & 31;
    int tile = blockIdx.x * 8 + wave;
    int coutBase = blockIdx.y * 16;
    int pixBase = tile * 32;
    // 32 consecutive pixels share (b, ho) since Wo is a multiple of 32
    int wo0 = pixBase % Wo;
    int t = pixBase / Wo;
    int ho = t % Ho;
    int b  = t / Ho;

    int row   = lane & 15;
    int khalf = (lane >> 4) << 4;   // 0 or 16
    v8f acc0 = {}, acc1 = {};
#pragma unroll
    for (int chk = 0; chk < NCH; ++chk) {
        const int kbase = chk * 32;
        // ---- im2col stage: lane owns k = kbase+lane for 32 pixels ----
        int k = kbase + lane;
        int ky = k / (3 * CINPAD);
        int r2 = k - ky * (3 * CINPAD);
        int kx = r2 / CINPAD;
        int ci = r2 - kx * CINPAD;
        int hi = ho * stride + ky - 1;
        bool rowOK = (ky < 3) && (hi >= 0) && (hi < H);
        const half_t* srcRow =
            act + (((size_t)b * H + (rowOK ? hi : 0)) * W) * CINPAD + ci;
        __builtin_prefetch(srcRow, 0, 1);   // global_prefetch_b8
#pragma unroll 4
        for (int m = 0; m < 32; ++m) {
            int wi = (wo0 + m) * stride + kx - 1;
            half_t v = (half_t)0.f;
            if (rowOK && wi >= 0 && wi < W) v = srcRow[(size_t)wi * CINPAD];
            ldsA[wave][m * 32 + lane] = v;
        }
        __syncthreads();
        // ---- fragments: two A tiles share one B fragment ----
        v16h a0 = *(const v16h*)&ldsA[wave][row * 32 + khalf];
        v16h a1 = *(const v16h*)&ldsA[wave][(row + 16) * 32 + khalf];
        v16h bf = *(const v16h*)&ldsW[(size_t)(coutBase + row) * KPAD + kbase + khalf];
        acc0 = __builtin_amdgcn_wmma_f32_16x16x32_f16(false, a0, false, bf,
                                                      (short)0, acc0, false, false);
        acc1 = __builtin_amdgcn_wmma_f32_16x16x32_f16(false, a1, false, bf,
                                                      (short)0, acc1, false, false);
        __syncthreads();
    }
    // ---- epilogue: bias, store, BN partial stats ----
    int n  = lane & 15;
    int mb = (lane >> 4) << 3;
    int cg = coutBase + n;
    float bv = bias ? bias[cg] : 0.f;
    float s = 0.f, s2 = 0.f;
#pragma unroll
    for (int r = 0; r < 8; ++r) {
        float v0 = acc0[r] + bv;
        float v1 = acc1[r] + bv;
        out[((size_t)pixBase + mb + r) * CoutPad + cg]      = v0;
        out[((size_t)pixBase + 16 + mb + r) * CoutPad + cg] = v1;
        s += v0 + v1; s2 += v0 * v0 + v1 * v1;
    }
    if (stats) {
        atomicAdd(&stats[cg], s);
        atomicAdd(&stats[CoutPad + cg], s2);
    }
}

__global__ void k_bn_stats(const float* __restrict__ stats, const float* __restrict__ g,
                           const float* __restrict__ be, float* __restrict__ scsh,
                           int Creal, int Cpad, float invc) {
    int c = threadIdx.x;
    if (c >= Cpad) return;
    float sc = 0.f, sh = 0.f;
    if (c < Creal) {
        float m = stats[c] * invc;
        float var = stats[Cpad + c] * invc - m * m;
        sc = g[c] * rsqrtf(var + 1e-5f);
        sh = be[c] - sc * m;
    }
    scsh[c] = sc; scsh[Cpad + c] = sh;
}

// BN(optional) + ReLU(optional) + skip-add(optional) + f32->f16
__global__ void k_bn_apply(const float* __restrict__ x, const float* __restrict__ scsh,
                           const half_t* __restrict__ skip, half_t* __restrict__ y,
                           size_t total, int Cpad, int Creal, int relu) {
    size_t i = (size_t)blockIdx.x * blockDim.x + threadIdx.x;
    if (i >= total) return;
    int c = (int)(i % Cpad);
    float v = 0.f;
    if (c < Creal) {
        v = x[i];
        if (scsh) v = v * scsh[c] + scsh[Cpad + c];
        if (relu) v = fmaxf(v, 0.f);
        if (skip) v += (float)skip[i];
    }
    y[i] = (half_t)v;
}

// bilinear 2x upsample (half-pixel centers), f16->f16, layout [B][H][W][C]
__global__ void k_up2(const half_t* __restrict__ in, half_t* __restrict__ out,
                      int H, int W, int C, size_t total) {
    size_t i = (size_t)blockIdx.x * blockDim.x + threadIdx.x;
    if (i >= total) return;
    int c = (int)(i % C); size_t t = i / C;
    int xo = (int)(t % (2 * W)); t /= (2 * W);
    int yo = (int)(t % (2 * H)); int b = (int)(t / (2 * H));
    float py = yo * 0.5f - 0.25f;
    float px = xo * 0.5f - 0.25f;
    int y0 = (int)floorf(py), x0 = (int)floorf(px);
    float fy = py - (float)y0, fx = px - (float)x0;
    int y1 = y0 + 1, x1 = x0 + 1;
    y0 = min(max(y0, 0), H - 1); y1 = min(max(y1, 0), H - 1);
    x0 = min(max(x0, 0), W - 1); x1 = min(max(x1, 0), W - 1);
    const half_t* base = in + (size_t)b * H * W * C + c;
    float s00 = (float)base[((size_t)y0 * W + x0) * C];
    float s01 = (float)base[((size_t)y0 * W + x1) * C];
    float s10 = (float)base[((size_t)y1 * W + x0) * C];
    float s11 = (float)base[((size_t)y1 * W + x1) * C];
    float v = (1.f - fy) * ((1.f - fx) * s00 + fx * s01)
            +        fy  * ((1.f - fx) * s10 + fx * s11);
    out[i] = (half_t)v;
}

// final 1x1 conv 8->3, write NCHW f32 to d_out
__global__ void k_final(const half_t* __restrict__ act, const float* __restrict__ wf,
                        float* __restrict__ out) {
    size_t i = (size_t)blockIdx.x * blockDim.x + threadIdx.x;  // over 32*16384
    if (i >= (size_t)32 * 16384) return;
    int b = (int)(i / 16384); int pix = (int)(i % 16384);
    const half_t* a = act + i * 16;
    for (int co = 0; co < 3; ++co) {
        float s = 0.f;
        for (int ci = 0; ci < 8; ++ci) s += (float)a[ci] * wf[co * 8 + ci];
        out[((size_t)b * 3 + co) * 16384 + pix] = s;
    }
}

// ============================================================
// Host driver
// ============================================================
extern "C" void kernel_launch(void* const* d_in, const int* in_sizes, int n_in,
                              void* d_out, int out_size, void* d_ws, size_t ws_size,
                              hipStream_t stream) {
    (void)in_sizes; (void)n_in; (void)out_size; (void)ws_size;
    const int B = 32;

    // ---- bump allocator over workspace ----
    char* wp = (char*)d_ws;
    auto alloc = [&](size_t bytes) -> void* {
        void* r = (void*)wp;
        wp += (bytes + 255) & ~(size_t)255;
        return r;
    };
    float*  xr    = (float*) alloc((size_t)B * 4 * 16384 * 4);
    float*  dsum  = (float*) alloc(256);
    float2* Frow  = (float2*)alloc((size_t)2 * B * 16384 * 8);
    float2* hat   = (float2*)alloc((size_t)B * 16384 * 8);
    float2* Gbuf  = (float2*)alloc((size_t)B * 16384 * 8);
    float*  ubuf  = (float*) alloc((size_t)B * 16384 * 4);
    half_t* actA  = (half_t*)alloc((size_t)B * 16384 * 32 * 2);
    half_t* actB  = (half_t*)alloc((size_t)B * 16384 * 32 * 2);
    half_t* b0buf = (half_t*)alloc((size_t)B * 16384 * 16 * 2);
    half_t* b1buf = (half_t*)alloc((size_t)B * 4096  * 16 * 2);
    float*  convO = (float*) alloc((size_t)B * 16384 * 16 * 4);
    half_t* wPack = (half_t*)alloc(32 * 288 * 2);
    float*  biasP = (float*) alloc(64 * 4);
    float*  stats = (float*) alloc(64 * 4);
    float*  scsh  = (float*) alloc(64 * 4);

    const float* x = (const float*)d_in[0];

    // ---- physics ----
    k_zero_f32<<<1, 32, 0, stream>>>(dsum, 1);
    k_down<<<(B * 4 * 16384) / 256, 256, 0, stream>>>(x, xr, dsum);
    k_rowdft<<<dim3(B * 128, 2), 128, 0, stream>>>(xr, Frow);
    k_colop<<<dim3(B, 128), 128, 0, stream>>>(Frow, hat, dsum);
    k_invcol<<<dim3(B, 128), 128, 0, stream>>>(hat, Gbuf);
    k_invrow<<<B * 128, 128, 0, stream>>>(Gbuf, ubuf);
    k_verlet<<<B, 1024, 4 * 16384 * 4, stream>>>(ubuf, xr, actA);  // actA = prop f16 [.,16]

    // ---- conv+BN helper ----
    auto run_conv = [&](const half_t* actIn, int H, int W, int Cin, int CinPad,
                        const float* wRaw, const float* bRaw,
                        const float* gRaw, const float* beRaw,
                        int Cout, int CoutPad, int stride,
                        const half_t* skip, half_t* actOut, int relu) {
        int Ho = H / stride, Wo = W / stride;
        int KPad = ((9 * CinPad) + 31) & ~31;
        int wn = CoutPad * KPad;
        k_pack_w<<<(wn + 255) / 256, 256, 0, stream>>>(wRaw, wPack, Cout, Cin, CinPad, CoutPad, KPad);
        if (bRaw) k_pack_vec<<<1, 64, 0, stream>>>(bRaw, biasP, Cout, CoutPad);
        bool bn = (gRaw != nullptr);
        if (bn) k_zero_f32<<<1, 64, 0, stream>>>(stats, 2 * CoutPad);
        dim3 grid((B * Ho * Wo / 32) / 8, CoutPad / 16);
        if (CinPad == 16)
            k_conv_wmma<16><<<grid, 256, 0, stream>>>(actIn, wPack, bRaw ? biasP : nullptr,
                                                      convO, bn ? stats : nullptr,
                                                      H, W, Ho, Wo, CoutPad, stride);
        else
            k_conv_wmma<32><<<grid, 256, 0, stream>>>(actIn, wPack, bRaw ? biasP : nullptr,
                                                      convO, bn ? stats : nullptr,
                                                      H, W, Ho, Wo, CoutPad, stride);
        size_t npix = (size_t)B * Ho * Wo;
        if (bn) k_bn_stats<<<1, CoutPad, 0, stream>>>(stats, gRaw, beRaw, scsh,
                                                      Cout, CoutPad, 1.f / (float)npix);
        size_t tot = npix * (size_t)CoutPad;
        k_bn_apply<<<(tot + 255) / 256, 256, 0, stream>>>(convO, bn ? scsh : nullptr,
                                                          skip, actOut, tot, CoutPad, Cout, relu);
    };
    auto F = [&](int i) { return (const float*)d_in[i]; };

    // ---- U-Net ----
    // db0: 4->8, 8->8 @128
    run_conv(actA, 128, 128, 4, 16, F(1), F(2), F(3), F(4), 8, 16, 1, nullptr, actB, 1);
    run_conv(actB, 128, 128, 8, 16, F(5), F(6), F(7), F(8), 8, 16, 1, nullptr, b0buf, 1);
    // dconv1: 8->16 stride 2 -> 64^2 (no bias/bn)
    run_conv(b0buf, 128, 128, 8, 16, F(9), nullptr, nullptr, nullptr, 16, 16, 2, nullptr, actA, 0);
    // db1: 16->16, 16->16 @64
    run_conv(actA, 64, 64, 16, 16, F(10), F(11), F(12), F(13), 16, 16, 1, nullptr, actB, 1);
    run_conv(actB, 64, 64, 16, 16, F(14), F(15), F(16), F(17), 16, 16, 1, nullptr, b1buf, 1);
    // dconv2: 16->32 stride 2 -> 32^2
    run_conv(b1buf, 64, 64, 16, 16, F(18), nullptr, nullptr, nullptr, 32, 32, 2, nullptr, actA, 0);
    // db2: 32->32 x2 @32
    run_conv(actA, 32, 32, 32, 32, F(19), F(20), F(21), F(22), 32, 32, 1, nullptr, actB, 1);
    run_conv(actB, 32, 32, 32, 32, F(23), F(24), F(25), F(26), 32, 32, 1, nullptr, actA, 1);
    // up2: 32ch 32^2 -> 64^2
    {
        size_t tot = (size_t)B * 4 * 32 * 32 * 32;
        k_up2<<<(tot + 255) / 256, 256, 0, stream>>>(actA, actB, 32, 32, 32, tot);
    }
    // up0: 32->16, 16->16 (+b1) @64
    run_conv(actB, 64, 64, 32, 32, F(27), F(28), F(29), F(30), 16, 16, 1, nullptr, actA, 1);
    run_conv(actA, 64, 64, 16, 16, F(31), F(32), F(33), F(34), 16, 16, 1, b1buf, actB, 1);
    // up2: 16ch 64^2 -> 128^2
    {
        size_t tot = (size_t)B * 4 * 64 * 64 * 16;
        k_up2<<<(tot + 255) / 256, 256, 0, stream>>>(actB, actA, 64, 64, 16, tot);
    }
    // up1: 16->8, 8->8 (+b0) @128
    run_conv(actA, 128, 128, 16, 16, F(35), F(36), F(37), F(38), 8, 16, 1, nullptr, actB, 1);
    run_conv(actB, 128, 128, 8, 16, F(39), F(40), F(41), F(42), 8, 16, 1, b0buf, actA, 1);
    // lastb: 8->8 x2 @128
    run_conv(actA, 128, 128, 8, 16, F(43), F(44), F(45), F(46), 8, 16, 1, nullptr, actB, 1);
    run_conv(actB, 128, 128, 8, 16, F(47), F(48), F(49), F(50), 8, 16, 1, nullptr, actA, 1);
    // final 1x1: 8->3, NCHW f32
    {
        size_t tot = (size_t)B * 16384;
        k_final<<<(tot + 255) / 256, 256, 0, stream>>>(actA, F(51), (float*)d_out);
    }
}
